// FEMBAFullTest_79182017069436
// MI455X (gfx1250) — matrix operand order repaired and checked by reading, not compiled, not run on hardware
//
#include <hip/hip_runtime.h>
#include <hip/hip_bf16.h>

typedef __attribute__((ext_vector_type(16))) _Float16 v16h;
typedef __attribute__((ext_vector_type(8)))  _Float16 v8h;
typedef __attribute__((ext_vector_type(8)))  float    v8f;
typedef __attribute__((ext_vector_type(4)))  unsigned int u32x4;
typedef __attribute__((ext_vector_type(4)))  int          i32x4;
typedef __attribute__((ext_vector_type(8)))  int          i32x8;

#define CDIV(a,b) (((a)+(b)-1)/(b))

#if defined(__has_builtin)
#if __has_builtin(__builtin_amdgcn_tensor_load_to_lds) && \
    __has_builtin(__builtin_amdgcn_s_wait_tensorcnt)
#define HAVE_TDM 1
#endif
#endif
#ifndef HAVE_TDM
#define HAVE_TDM 0
#endif

// ---------------------------------------------------------------- fq helpers
__global__ void fq_zero_k(unsigned* slot) {
    if (threadIdx.x == 0 && blockIdx.x == 0) *slot = 0u;
}

__global__ void fq_absmax_k(const float* __restrict__ x, size_t n, unsigned* slot) {
    float m = 0.f;
    for (size_t i = (size_t)blockIdx.x * blockDim.x + threadIdx.x; i < n;
         i += (size_t)gridDim.x * blockDim.x)
        m = fmaxf(m, fabsf(x[i]));
    for (int o = 16; o > 0; o >>= 1) m = fmaxf(m, __shfl_xor(m, o, 32));
    __shared__ float sm[8];
    int lane = threadIdx.x & 31, wid = threadIdx.x >> 5;
    if (lane == 0) sm[wid] = m;
    __syncthreads();
    if (wid == 0) {
        m = (lane < (int)(blockDim.x >> 5)) ? sm[lane] : 0.f;
        for (int o = 4; o > 0; o >>= 1) m = fmaxf(m, __shfl_xor(m, o, 32));
        if (lane == 0) atomicMax(slot, __float_as_uint(m));
    }
}

__global__ void fq_apply_k(float* __restrict__ x, size_t n, const unsigned* slot) {
    float s = fmaxf(__uint_as_float(*slot), 1e-8f) / 127.f;
    float inv = 1.f / s;
    for (size_t i = (size_t)blockIdx.x * blockDim.x + threadIdx.x; i < n;
         i += (size_t)gridDim.x * blockDim.x) {
        float q = rintf(x[i] * inv);
        q = fminf(fmaxf(q, -128.f), 127.f);
        x[i] = q * s;
    }
}

// ---------------------------------------------------------------- elementwise utils
__global__ void cvt_f16_k(const float* __restrict__ x, _Float16* __restrict__ y, size_t n) {
    for (size_t i = (size_t)blockIdx.x * blockDim.x + threadIdx.x; i < n;
         i += (size_t)gridDim.x * blockDim.x)
        y[i] = (_Float16)x[i];
}

// WT[n*K+k] = W[k*N+n], converted to f16
__global__ void convT_f16_k(const float* __restrict__ W, _Float16* __restrict__ WT,
                            int K, int N) {
    size_t n = (size_t)K * N;
    for (size_t i = (size_t)blockIdx.x * blockDim.x + threadIdx.x; i < n;
         i += (size_t)gridDim.x * blockDim.x) {
        int k = (int)(i / N), c = (int)(i % N);
        WT[(size_t)c * K + k] = (_Float16)W[i];
    }
}

__global__ void add_pos_k(float* __restrict__ h, const float* __restrict__ pos,
                          size_t n, size_t posN) {
    for (size_t i = (size_t)blockIdx.x * blockDim.x + threadIdx.x; i < n;
         i += (size_t)gridDim.x * blockDim.x)
        h[i] += pos[i % posN];
}

// ---------------------------------------------------------------- patch embed
__global__ void patch_k(const float* __restrict__ x, const float* __restrict__ pw,
                        const float* __restrict__ pb, float* __restrict__ h,
                        int B, int H, int W, int L, int D, int EMB) {
    size_t n = (size_t)B * L * D;
    for (size_t i = (size_t)blockIdx.x * blockDim.x + threadIdx.x; i < n;
         i += (size_t)gridDim.x * blockDim.x) {
        int d = (int)(i % D);
        size_t bl = i / D;
        int l = (int)(bl % L);
        int b = (int)(bl / L);
        int gy = d / EMB, c = d % EMB;
        float acc = pb[c];
        for (int ph = 0; ph < 2; ph++)
            for (int pq = 0; pq < 2; pq++)
                acc += x[((size_t)b * H + (2 * gy + ph)) * W + (2 * l + pq)] *
                       pw[c * 4 + ph * 2 + pq];
        h[i] = acc;
    }
}

// ---------------------------------------------------------------- WMMA GEMM (LDS-staged A)
// C[M,N] = A[M,K=256](f16) * BT[N,K=256](f16)^T, f32 accumulate.
// One block = 8 waves sharing a 16-row A panel staged in LDS (TDM async load
// when available, padded 528B/row to spread LDS banks). Each wave owns a
// 16x32 slice of C; block covers 16 x 256 of C.
#define GEMM_K 256
#define LDS_ROW 264   // 256 halves + 4-DWORD pad = 528 bytes

__global__ void gemm_wmma_f16_k(const _Float16* __restrict__ A,
                                const _Float16* __restrict__ BT,
                                float* __restrict__ C,
                                int M, int N, int K, int accumulate) {
    __shared__ _Float16 smem[16 * LDS_ROW];
    const int lane = threadIdx.x;          // 0..31
    const int wave = threadIdx.y;          // 0..7
    const int mIdx = blockIdx.x;           // M/16 tiles
    const int nChunk = blockIdx.y * 8 + wave;  // 16x32 tile index along N

    // ---- stage A panel [16 x 256] into LDS ----
#if HAVE_TDM
    if (wave == 0) {
        unsigned long long ga =
            (unsigned long long)(const void*)(A + (size_t)mIdx * 16 * K);
        u32x4 g0;
        g0[0] = 1u;                                      // count=1, user mode
        g0[1] = 0u;                                      // lds_addr (smem @ 0)
        g0[2] = (unsigned)(ga & 0xFFFFFFFFu);            // global_addr[31:0]
        g0[3] = (unsigned)((ga >> 32) & 0x1FFFFFFu) | (2u << 30); // addr hi | type=2
        i32x8 g1;
        g1[0] = (1 << 16) | (1 << 20) | (6 << 22) | (3 << 25);
        //        data=2B   pad_en      every 128dw  pad 4dw (528B rows)
        g1[1] = (int)(K << 16);   // tensor_dim0[15:0] = 256
        g1[2] = (16 << 16);       // tensor_dim1[15:0] = 16 rows
        g1[3] = (int)(K << 16);   // tile_dim0 = 256
        g1[4] = 16;               // tile_dim1 = 16
        g1[5] = K;                // tensor_dim0_stride = K elements
        g1[6] = 0; g1[7] = 0;
        i32x4 gz = {0, 0, 0, 0};
#if __clang_major__ >= 23
        i32x8 gz8 = {0, 0, 0, 0, 0, 0, 0, 0};
        __builtin_amdgcn_tensor_load_to_lds(g0, g1, gz, gz, gz8, 0);
#else
        __builtin_amdgcn_tensor_load_to_lds(g0, g1, gz, gz, 0);
#endif
        __builtin_amdgcn_s_wait_tensorcnt(0);
    }
#else
    {
        int t = wave * 32 + lane;          // 0..255
        int r = t >> 4, chunk = t & 15;    // 16 rows x 16 chunks of 16 halves
        const _Float16* src = A + ((size_t)(mIdx * 16 + r)) * K + chunk * 16;
        _Float16* dst = smem + r * LDS_ROW + chunk * 16;
        *(v8h*)dst       = *(const v8h*)src;
        *(v8h*)(dst + 8) = *(const v8h*)(src + 8);
    }
#endif
    __syncthreads();
    // The TDM engine writes LDS behind the compiler's back (the builtin takes
    // no pointer to smem). smem's address must ESCAPE into the asm, otherwise
    // capture analysis proves the clobber can't touch it and every A-fragment
    // load folds to undef (round-2/3 asm: ds_loads gone, A aliased onto an
    // accumulator, 5x v_nop hazard stalls between WMMAs).
    {
        _Float16* smem_escape = smem;
        asm volatile("" : "+r"(smem_escape) : : "memory");
    }

    if (nChunk * 32 >= N) return;          // idle wave (after barrier)

    const int row16 = lane & 15;
    const int col0  = nChunk * 32 + (lane & 15);
    const int col1  = col0 + 16;
    const int kbA = (lane >> 4) * 16;      // A frag: K = kbA..+15
    const int kbB = (lane >> 4) * 16;      // B frag: K = kbB..+15

    v8f acc0 = {}, acc1 = {};
    for (int k = 0; k < K; k += 32) {
        v8h a_lo = *(const v8h*)(smem + row16 * LDS_ROW + k + kbA);
        v8h a_hi = *(const v8h*)(smem + row16 * LDS_ROW + k + kbA + 8);
        v16h b0  = *(const v16h*)(BT + (size_t)col0 * K + k + kbB);
        v16h b1  = *(const v16h*)(BT + (size_t)col1 * K + k + kbB);
        v16h a;
        #pragma unroll
        for (int e = 0; e < 8; e++) { a[e] = a_lo[e]; a[e + 8] = a_hi[e]; }
        acc0 = __builtin_amdgcn_wmma_f32_16x16x32_f16(false, a, false, b0,
                                                      (short)0, acc0, false, false);
        acc1 = __builtin_amdgcn_wmma_f32_16x16x32_f16(false, a, false, b1,
                                                      (short)0, acc1, false, false);
    }
    int orow = mIdx * 16 + (lane >> 4) * 8;
    int ocol = nChunk * 32 + (lane & 15);
    #pragma unroll
    for (int r = 0; r < 8; r++) {
        size_t i0 = (size_t)(orow + r) * N + ocol;
        if (accumulate) { C[i0] += acc0[r]; C[i0 + 16] += acc1[r]; }
        else            { C[i0]  = acc0[r]; C[i0 + 16]  = acc1[r]; }
    }
}

// ---------------------------------------------------------------- depthwise conv + SiLU
__global__ void conv_silu_k(const float* __restrict__ xz, const float* __restrict__ w,
                            const float* __restrict__ bias, float* __restrict__ xs,
                            _Float16* __restrict__ xs16, int B, int L, int DI, int dir) {
    size_t n = (size_t)B * L * DI;
    for (size_t i = (size_t)blockIdx.x * blockDim.x + threadIdx.x; i < n;
         i += (size_t)gridDim.x * blockDim.x) {
        int d = (int)(i % DI);
        size_t bl = i / DI;
        int l = (int)(bl % L);
        int b = (int)(bl / L);
        float acc = bias[d];
        #pragma unroll
        for (int k = 0; k < 4; k++) {
            int ls = l + (dir ? (3 - k) : (k - 3));
            if (ls >= 0 && ls < L)
                acc += w[d * 4 + k] * xz[((size_t)b * L + ls) * 512 + d];
        }
        float s = acc / (1.f + __expf(-acc));
        xs[i] = s;
        xs16[i] = (_Float16)s;
    }
}

// ---------------------------------------------------------------- dt = softplus(dt_in @ W_dt + b_dt)
__global__ void dt_k(const float* __restrict__ proj, const float* __restrict__ Wdt,
                     const float* __restrict__ bdt, float* __restrict__ dt,
                     size_t Mr, int DI) {
    size_t n = Mr * DI;
    for (size_t i = (size_t)blockIdx.x * blockDim.x + threadIdx.x; i < n;
         i += (size_t)gridDim.x * blockDim.x) {
        int d = (int)(i % DI);
        size_t m = i / DI;
        float acc = bdt[d];
        #pragma unroll
        for (int r = 0; r < 16; r++)
            acc += proj[m * 32 + r] * Wdt[r * DI + d];
        dt[i] = (acc > 20.f) ? acc : log1pf(__expf(acc));
    }
}

// ---------------------------------------------------------------- selective scan
__global__ void scan_k(const float* __restrict__ dt, const float* __restrict__ proj,
                       const float* __restrict__ xs, const float* __restrict__ xz,
                       const float* __restrict__ A_log, const float* __restrict__ Dsk,
                       float* __restrict__ ypre, int B, int L, int DI, int dir) {
    int g = blockIdx.x * blockDim.x + threadIdx.x;   // b*DI + d, d fastest
    if (g >= B * DI) return;
    int b = g / DI, d = g % DI;
    float A[8], h[8];
    #pragma unroll
    for (int s = 0; s < 8; s++) { A[s] = -__expf(A_log[d * 8 + s]); h[s] = 0.f; }
    float Dv = Dsk[d];
    for (int t = 0; t < L; t++) {
        int l = dir ? (L - 1 - t) : t;
        size_t base = (size_t)b * L + l;
        float dtv = dt[base * DI + d];
        float xv  = xs[base * DI + d];
        float zv  = xz[base * 512 + 256 + d];
        float dx  = dtv * xv;
        float y = 0.f;
        #pragma unroll
        for (int s = 0; s < 8; s++) {
            float dA = __expf(dtv * A[s]);
            h[s] = dA * h[s] + dx * proj[base * 32 + 16 + s];
            y += h[s] * proj[base * 32 + 24 + s];
        }
        float gate = zv / (1.f + __expf(-zv));
        ypre[base * DI + d] = (y + xv * Dv) * gate;
    }
}

// ---------------------------------------------------------------- layernorm(res + y)
__global__ void ln_k(float* __restrict__ h, const float* __restrict__ y,
                     const float* __restrict__ g, const float* __restrict__ bt,
                     size_t rows, int D) {
    int lane = threadIdx.x;
    size_t row = (size_t)blockIdx.x * blockDim.y + threadIdx.y;
    if (row >= rows) return;
    const float* hp = h + row * D;
    const float* yp = y + row * D;
    float vals[8], sum = 0.f, sum2 = 0.f;
    #pragma unroll
    for (int j = 0; j < 8; j++) {
        float v = hp[lane + j * 32] + yp[lane + j * 32];
        vals[j] = v; sum += v; sum2 += v * v;
    }
    for (int o = 16; o > 0; o >>= 1) {
        sum  += __shfl_xor(sum,  o, 32);
        sum2 += __shfl_xor(sum2, o, 32);
    }
    float mean = sum / D;
    float var  = sum2 / D - mean * mean;
    float inv  = rsqrtf(var + 1e-5f);
    #pragma unroll
    for (int j = 0; j < 8; j++) {
        int c = lane + j * 32;
        h[row * D + c] = (vals[j] - mean) * inv * g[c] + bt[c];
    }
}

// ---------------------------------------------------------------- pool + classifier
__global__ void pool_k(const float* __restrict__ h, float* __restrict__ pooled,
                       int B, int L, int D) {
    int i = blockIdx.x * blockDim.x + threadIdx.x;
    if (i >= B * D) return;
    int b = i / D, d = i % D;
    float s = 0.f;
    for (int l = 0; l < L; l++) s += h[((size_t)b * L + l) * D + d];
    pooled[i] = s / (float)L;
}

__global__ void cls_k(const float* __restrict__ pooled, const float* __restrict__ wq,
                      const float* __restrict__ cb, float* __restrict__ out,
                      int B, int NC, int D) {
    int i = blockIdx.x * blockDim.x + threadIdx.x;
    if (i >= B * NC) return;
    int b = i / NC, n = i % NC;
    float acc = cb[n];
    for (int d = 0; d < D; d++) acc += pooled[b * D + d] * wq[n * D + d];
    out[i] = acc;
}

// ================================================================ host
extern "C" void kernel_launch(void* const* d_in, const int* in_sizes, int n_in,
                              void* d_out, int out_size, void* d_ws, size_t ws_size,
                              hipStream_t stream) {
    (void)in_sizes; (void)n_in; (void)out_size; (void)ws_size;
    const float* x_in    = (const float*)d_in[0];
    const float* patch_w = (const float*)d_in[1];
    const float* patch_b = (const float*)d_in[2];
    const float* pos_e   = (const float*)d_in[3];
    const float* W_in    = (const float*)d_in[4];
    const float* conv_w  = (const float*)d_in[5];
    const float* conv_b  = (const float*)d_in[6];
    const float* W_xp    = (const float*)d_in[7];
    const float* W_dt    = (const float*)d_in[8];
    const float* b_dt    = (const float*)d_in[9];
    const float* A_log   = (const float*)d_in[10];
    const float* D_skip  = (const float*)d_in[11];
    const float* W_out   = (const float*)d_in[12];
    const float* ln_g    = (const float*)d_in[13];
    const float* ln_b    = (const float*)d_in[14];
    const float* cls_w   = (const float*)d_in[15];
    const float* cls_b   = (const float*)d_in[16];
    float* out = (float*)d_out;

    constexpr int B = 32, H = 16, W = 4096, L = 2048, D = 256, DI = 256;
    constexpr int EMB = 32, NB = 2, DTR = 16, DS = 8, NC = 10;
    const size_t M = (size_t)B * L;

    char* p = (char*)d_ws;
    auto alloc = [&](size_t bytes) -> void* {
        void* r = (void*)p;
        p += (bytes + 255) & ~(size_t)255;
        return r;
    };
    float*     bufH   = (float*)alloc(M * D * 4);
    float*     bufPos = (float*)alloc((size_t)L * D * 4);
    float*     xq     = (float*)alloc((size_t)B * H * W * 4);
    float*     pwq    = (float*)alloc(EMB * 4 * 4);
    _Float16*  hF16   = (_Float16*)alloc(M * D * 2);
    float*     bufXZ  = (float*)alloc(M * 512 * 4);
    float*     bufXS  = (float*)alloc(M * DI * 4);
    _Float16*  xsF16  = (_Float16*)alloc(M * DI * 2);
    float*     bufPr  = (float*)alloc(M * 32 * 4);
    float*     bufDT  = (float*)alloc(M * DI * 4);
    float*     bufYp  = (float*)alloc(M * DI * 4);
    _Float16*  ypF16  = (_Float16*)alloc(M * DI * 2);
    float*     bufY   = (float*)alloc(M * D * 4);
    _Float16*  wtIn   = (_Float16*)alloc((size_t)512 * 256 * 2);
    _Float16*  wtXp   = (_Float16*)alloc((size_t)32 * 256 * 2);
    _Float16*  wtOut  = (_Float16*)alloc((size_t)256 * 256 * 2);
    float*     clsQ   = (float*)alloc((size_t)NC * D * 4);
    float*     pooled = (float*)alloc((size_t)B * D * 4);
    unsigned*  slot   = (unsigned*)alloc(256);

    auto egrid = [](size_t n) { return (int)CDIV(n, (size_t)256); };
    auto fq = [&](float* ptr, size_t n) {
        fq_zero_k<<<1, 1, 0, stream>>>(slot);
        int rb = (int)CDIV(n, (size_t)256); if (rb > 2048) rb = 2048;
        fq_absmax_k<<<rb, 256, 0, stream>>>(ptr, n, slot);
        fq_apply_k<<<egrid(n), 256, 0, stream>>>(ptr, n, slot);
    };
    auto gemm = [&](const _Float16* A, const _Float16* BT, float* C,
                    int Mm, int Nn, int Kk, int acc) {
        dim3 blk(32, 8);
        dim3 grd(Mm / 16, CDIV(Nn / 32, 8));
        gemm_wmma_f16_k<<<grd, blk, 0, stream>>>(A, BT, C, Mm, Nn, Kk, acc);
    };

    // ---- input fq + patch embed + pos ----
    hipMemcpyAsync(xq, x_in, (size_t)B * H * W * 4, hipMemcpyDeviceToDevice, stream);
    fq(xq, (size_t)B * H * W);
    hipMemcpyAsync(pwq, patch_w, EMB * 4 * 4, hipMemcpyDeviceToDevice, stream);
    fq(pwq, EMB * 4);
    patch_k<<<egrid(M * D), 256, 0, stream>>>(xq, pwq, patch_b, bufH, B, H, W, L, D, EMB);
    fq(bufH, M * D);
    hipMemcpyAsync(bufPos, pos_e, (size_t)L * D * 4, hipMemcpyDeviceToDevice, stream);
    fq(bufPos, (size_t)L * D);
    fq(bufH, M * D);
    fq(bufPos, (size_t)L * D);
    add_pos_k<<<egrid(M * D), 256, 0, stream>>>(bufH, bufPos, M * D, (size_t)L * D);
    fq(bufH, M * D);

    // ---- mamba blocks ----
    for (int i = 0; i < NB; i++) {
        cvt_f16_k<<<egrid(M * D), 256, 0, stream>>>(bufH, hF16, M * D);
        for (int dir = 0; dir < 2; dir++) {
            const int bd = i * 2 + dir;
            convT_f16_k<<<egrid((size_t)D * 512), 256, 0, stream>>>(
                W_in + (size_t)bd * D * 512, wtIn, D, 512);
            gemm(hF16, wtIn, bufXZ, (int)M, 512, D, 0);                 // xz
            conv_silu_k<<<egrid(M * DI), 256, 0, stream>>>(
                bufXZ, conv_w + (size_t)bd * DI * 4, conv_b + (size_t)bd * DI,
                bufXS, xsF16, B, L, DI, dir);
            convT_f16_k<<<egrid((size_t)DI * 32), 256, 0, stream>>>(
                W_xp + (size_t)bd * DI * 32, wtXp, DI, 32);
            gemm(xsF16, wtXp, bufPr, (int)M, 32, DI, 0);                // proj
            dt_k<<<egrid(M * DI), 256, 0, stream>>>(
                bufPr, W_dt + (size_t)bd * DTR * DI, b_dt + (size_t)bd * DI,
                bufDT, M, DI);
            scan_k<<<CDIV(B * DI, 256), 256, 0, stream>>>(
                bufDT, bufPr, bufXS, bufXZ, A_log + (size_t)bd * DI * DS,
                D_skip + (size_t)bd * DI, bufYp, B, L, DI, dir);
            cvt_f16_k<<<egrid(M * DI), 256, 0, stream>>>(bufYp, ypF16, M * DI);
            convT_f16_k<<<egrid((size_t)DI * D), 256, 0, stream>>>(
                W_out + (size_t)bd * DI * D, wtOut, DI, D);
            gemm(ypF16, wtOut, bufY, (int)M, D, DI, dir);               // += y@W_out
        }
        fq(bufY, M * D);
        ln_k<<<dim3((unsigned)CDIV(M, (size_t)8)), dim3(32, 8), 0, stream>>>(
            bufH, bufY, ln_g + (size_t)i * D, ln_b + (size_t)i * D, M, D);
        fq(bufH, M * D);
    }

    // ---- pool + classifier ----
    pool_k<<<CDIV(B * D, 256), 256, 0, stream>>>(bufH, pooled, B, L, D);
    fq(pooled, (size_t)B * D);
    hipMemcpyAsync(clsQ, cls_w, (size_t)NC * D * 4, hipMemcpyDeviceToDevice, stream);
    fq(clsQ, (size_t)NC * D);
    cls_k<<<CDIV(B * NC, 256), 256, 0, stream>>>(pooled, clsQ, cls_b, out, B, NC, D);
}
